// Qwen3VL_81801947120308
// MI455X (gfx1250) — matrix operand order repaired and checked by reading, not compiled, not run on hardware
//
#include <hip/hip_runtime.h>
#include <hip/hip_bf16.h>
#include <cstdint>
#include <cstddef>

// ---------------- model constants ----------------
#define L_  2
#define B_  2
#define T_  1024
#define C_  2048
#define H_  32
#define KV_ 4
#define DH_ 128
#define E_  8
#define M_  768
#define BT_ (B_*T_)
#define HD_ (H_*DH_)    // 4096
#define KVD_ (KV_*DH_)  // 512

typedef unsigned short u16;
typedef __attribute__((ext_vector_type(16))) __bf16 bf16x16;
typedef __attribute__((ext_vector_type(8)))  float  f32x8;
typedef __attribute__((ext_vector_type(4)))  unsigned v4u;
typedef __attribute__((ext_vector_type(4)))  int      v4i;
typedef __attribute__((ext_vector_type(8)))  int      v8i;

union FragA { bf16x16 v; uint4 q[2]; unsigned w[8]; u16 s[16]; };
union FragC { f32x8 v; float f[8]; };

static __device__ __forceinline__ u16 f2bf(float f) {
    union { float f; unsigned u; } x; x.f = f;
    unsigned r = x.u + 0x7FFFu + ((x.u >> 16) & 1u);
    return (u16)(r >> 16);
}
static __device__ __forceinline__ float bf2f(u16 h) {
    union { unsigned u; float f; } x; x.u = ((unsigned)h) << 16;
    return x.f;
}

// ---------------------------------------------------------------------
// TDM: 2D bf16 tensor-tile load global -> LDS with pad so a 32-elem row
// lands with stride 40 ushorts (pad_interval=16dw, pad_amount=4dw).
// ---------------------------------------------------------------------
static __device__ __forceinline__ void tdm_load_2d_bf16(
    unsigned lds_off, const u16* gsrc,
    unsigned K_elems, unsigned N_rows,
    unsigned tile_k, unsigned tile_rows)
{
    unsigned long long ga = (unsigned long long)(uintptr_t)gsrc;
    v4u g0;
    g0[0] = 1u;                                  // count=1 (valid user D#)
    g0[1] = lds_off;                             // lds_addr (bytes)
    g0[2] = (unsigned)(ga & 0xFFFFFFFFu);        // global_addr[31:0]
    g0[3] = (unsigned)((ga >> 32) & 0x01FFFFFFu) | (2u << 30);  // addr[56:32], type=2
    v8i g1;
    g1[0] = (int)((1u << 16) | (1u << 20) | (3u << 22) | (3u << 25));
    g1[1] = (int)((K_elems & 0xFFFFu) << 16);
    g1[2] = (int)(((K_elems >> 16) & 0xFFFFu) | ((N_rows & 0xFFFFu) << 16));
    g1[3] = (int)(((N_rows >> 16) & 0xFFFFu) | ((tile_k & 0xFFFFu) << 16));
    g1[4] = (int)(tile_rows & 0xFFFFu);
    g1[5] = (int)K_elems;
    g1[6] = 0;
    g1[7] = 0;
    v4i gz = {0, 0, 0, 0};
#if __has_include(<hip/amd_detail/amd_gfx1250_TDM.h>)
    v8i gz8 = {0, 0, 0, 0, 0, 0, 0, 0};
    __builtin_amdgcn_tensor_load_to_lds(g0, g1, gz, gz, gz8, 0);
#else
    __builtin_amdgcn_tensor_load_to_lds(g0, g1, gz, gz, 0);
#endif
}

__global__ void cvt_k(const float* __restrict__ src, u16* __restrict__ dst, size_t n)
{
    size_t i = (size_t)blockIdx.x * 256 + threadIdx.x;
    if (i < n) dst[i] = f2bf(src[i]);
}

// =====================================================================
// GEMM with optional token routing:
//   A rows optionally gathered through gIdx[slot];
//   effective M read from *cntPtr (blocks past it exit);
//   output rows optionally scattered via sIdx[slot], scaled by
//   slotScale[slot], with optional += (beta).
// bf16 WMMA 16x16x32, block tile 64x128xK32, weights staged by TDM,
// double buffered; one barrier + one s_wait_tensorcnt per K-step.
// =====================================================================
__global__ __launch_bounds__(256) void gemm_bf16_wmma(
    const u16* __restrict__ A, const u16* __restrict__ W,
    float* __restrict__ Out,
    int M, int N, int K, int beta,
    const int* __restrict__ gIdx, const int* __restrict__ sIdx,
    const float* __restrict__ slotScale, const int* __restrict__ cntPtr)
{
    __shared__ __align__(16) u16 As[2][64 * 40];
    __shared__ __align__(16) u16 Bs[2][128 * 40];

    const int tid  = threadIdx.x;
    const int lane = tid & 31, wid = tid >> 5;
    const int mBase = blockIdx.y * 64, nBase = blockIdx.x * 128;
    const int Meff = cntPtr ? *cntPtr : M;
    if (mBase >= Meff) return;

    const int mt = wid & 3, ng = (wid >> 2) * 4;
    const int hbA = (lane & 16) ? 8 : 0;
    const int hbB = (lane & 16) ? 8 : 0;
    const int r15 = lane & 15;
    const int srow = tid >> 2;              // 0..63
    const int scol = (tid & 3) * 8;         // 0,8,16,24

    const unsigned ldsA0 = (unsigned)(uintptr_t)(&As[0][0]);
    const unsigned ldsA1 = (unsigned)(uintptr_t)(&As[1][0]);
    const unsigned ldsB0 = (unsigned)(uintptr_t)(&Bs[0][0]);
    const unsigned ldsB1 = (unsigned)(uintptr_t)(&Bs[1][0]);

    FragC c[4];
    for (int t = 0; t < 4; ++t)
        for (int j = 0; j < 8; ++j) c[t].f[j] = 0.f;

    // gather row index for this thread's staging slice
    int tokRow = 0;
    if (gIdx) {
        int slot = mBase + srow;
        tokRow = (slot < Meff) ? gIdx[slot] : gIdx[mBase];
    }

    const int nk = K >> 5;
    // ---- stage K-tile 0 ----
    if (gIdx) {
        *(uint4*)&As[0][srow * 40 + scol] =
            *(const uint4*)(A + (size_t)tokRow * K + scol);
    } else if (wid == 0) {
        tdm_load_2d_bf16(ldsA0, A + (size_t)mBase * K, K, M, 32, 64);
    }
    if (wid == 0) {
        tdm_load_2d_bf16(ldsB0, W + (size_t)nBase * K, K, N, 32, 128);
        __builtin_amdgcn_s_wait_tensorcnt(0);
    }
    __syncthreads();

    int cur = 0;
    for (int kt = 0; kt < nk; ++kt) {
        const int nxt = cur ^ 1;
        if (kt + 1 < nk) {
            const int k1 = (kt + 1) << 5;
            if (gIdx) {
                *(uint4*)&As[nxt][srow * 40 + scol] =
                    *(const uint4*)(A + (size_t)tokRow * K + k1 + scol);
            } else if (wid == 0) {
                tdm_load_2d_bf16(nxt ? ldsA1 : ldsA0, A + (size_t)mBase * K + k1, K, M, 32, 64);
            }
            if (wid == 0)
                tdm_load_2d_bf16(nxt ? ldsB1 : ldsB0, W + (size_t)nBase * K + k1, K, N, 32, 128);
        }

        // batched fragment loads, then back-to-back WMMAs
        FragA fa, fb[4];
        const int ar = mt * 16 + r15;
        fa.q[0] = *(const uint4*)&As[cur][ar * 40 + hbA];
        fa.q[1] = *(const uint4*)&As[cur][ar * 40 + 16 + hbA];
        for (int t = 0; t < 4; ++t) {
            const int br = (ng + t) * 16 + r15;
            fb[t].q[0] = *(const uint4*)&Bs[cur][br * 40 + hbB];
            fb[t].q[1] = *(const uint4*)&Bs[cur][br * 40 + hbB + 16];
        }
        for (int t = 0; t < 4; ++t)
            c[t].v = __builtin_amdgcn_wmma_f32_16x16x32_bf16(false, fa.v, false, fb[t].v,
                                                             (short)0, c[t].v, false, false);

        if (wid == 0) __builtin_amdgcn_s_wait_tensorcnt(0);
        __syncthreads();
        cur = nxt;
    }

    for (int t = 0; t < 4; ++t)
        for (int j = 0; j < 8; ++j) {
            const int slot = mBase + mt * 16 + j + hbA;
            if (slot >= Meff) continue;
            const int orow = sIdx ? sIdx[slot] : slot;
            const int col  = nBase + (ng + t) * 16 + r15;
            float sc = slotScale ? slotScale[slot] : 1.0f;
            float v  = c[t].f[j] * sc;
            size_t o = (size_t)orow * N + col;
            Out[o] = beta ? Out[o] + v : v;
        }
}

// =====================================================================
// Fused causal GQA flash-attention with WMMA (bf16 in, fp32 softmax).
// =====================================================================
__global__ __launch_bounds__(128) void attn_kernel(
    const u16* __restrict__ Qb, const u16* __restrict__ Kb,
    const u16* __restrict__ Vb, u16* __restrict__ Y)
{
    __shared__ __align__(16) u16 Klds[32 * 136];
    __shared__ __align__(16) u16 Vt[128 * 40];
    __shared__ __align__(16) u16 Plds[4 * 16 * 40];

    const int tid = threadIdx.x, lane = tid & 31, w = tid >> 5;
    const int b = blockIdx.z, h = blockIdx.y;
    const int q0 = blockIdx.x * 64 + w * 16;
    const int kvh = h / (H_ / KV_);
    const int hbA = (lane & 16) ? 8 : 0;
    const int hbB = (lane & 16) ? 8 : 0;
    const int r15 = lane & 15;

    FragA qa[4];
    const u16* qrow = Qb + (((size_t)b * H_ + h) * T_ + (q0 + r15)) * DH_;
    for (int c = 0; c < 4; ++c) {
        qa[c].q[0] = *(const uint4*)(qrow + c * 32 + hbA);
        qa[c].q[1] = *(const uint4*)(qrow + c * 32 + 16 + hbA);
    }

    FragC acc[8];
    float m_i[8], l_i[8];
    for (int nt = 0; nt < 8; ++nt)
        for (int j = 0; j < 8; ++j) acc[nt].f[j] = 0.f;
    for (int j = 0; j < 8; ++j) { m_i[j] = -1e30f; l_i[j] = 0.f; }

    const float smscale = 0.08838834764831845f; // 1/sqrt(128)
    const int nIter = blockIdx.x * 2 + 2;

    for (int kb = 0; kb < nIter; ++kb) {
        const u16* kbase = Kb + (((size_t)b * KV_ + kvh) * T_ + kb * 32) * DH_;
        const u16* vbase = Vb + (((size_t)b * KV_ + kvh) * T_ + kb * 32) * DH_;
        for (int i = 0; i < 4; ++i) {
            int flat = tid * 8 + i * 1024;
            int row = flat >> 7, col = flat & 127;
            *(uint4*)&Klds[row * 136 + col] = *(const uint4*)(kbase + (size_t)row * DH_ + col);
            union { uint4 q; u16 s[8]; } vv;
            vv.q = *(const uint4*)(vbase + (size_t)row * DH_ + col);
            for (int e = 0; e < 8; ++e) Vt[(col + e) * 40 + row] = vv.s[e];
        }
        __syncthreads();

        if (kb * 32 <= q0 + 15) {
            FragC s0, s1;
            for (int j = 0; j < 8; ++j) { s0.f[j] = 0.f; s1.f[j] = 0.f; }
            for (int c = 0; c < 4; ++c) {
                FragA kb0, kb1;
                kb0.q[0] = *(const uint4*)&Klds[r15 * 136 + c * 32 + hbB];
                kb0.q[1] = *(const uint4*)&Klds[r15 * 136 + c * 32 + hbB + 16];
                kb1.q[0] = *(const uint4*)&Klds[(16 + r15) * 136 + c * 32 + hbB];
                kb1.q[1] = *(const uint4*)&Klds[(16 + r15) * 136 + c * 32 + hbB + 16];
                s0.v = __builtin_amdgcn_wmma_f32_16x16x32_bf16(false, qa[c].v, false, kb0.v,
                                                               (short)0, s0.v, false, false);
                s1.v = __builtin_amdgcn_wmma_f32_16x16x32_bf16(false, qa[c].v, false, kb1.v,
                                                               (short)0, s1.v, false, false);
            }
            float mb[8], rs[8];
            for (int j = 0; j < 8; ++j) {
                int q = q0 + j + hbA;
                s0.f[j] *= smscale; s1.f[j] *= smscale;
                if (kb * 32 + r15 > q)      s0.f[j] = -1e30f;
                if (kb * 32 + 16 + r15 > q) s1.f[j] = -1e30f;
                mb[j] = fmaxf(s0.f[j], s1.f[j]);
            }
            for (int m = 1; m < 16; m <<= 1)
                for (int j = 0; j < 8; ++j)
                    mb[j] = fmaxf(mb[j], __shfl_xor(mb[j], m, 32));
            for (int j = 0; j < 8; ++j) {
                float mn  = fmaxf(m_i[j], mb[j]);
                float scl = expf(m_i[j] - mn);
                l_i[j] *= scl;
                for (int nt = 0; nt < 8; ++nt) acc[nt].f[j] *= scl;
                s0.f[j] = expf(s0.f[j] - mn);
                s1.f[j] = expf(s1.f[j] - mn);
                rs[j] = s0.f[j] + s1.f[j];
                m_i[j] = mn;
            }
            for (int m = 1; m < 16; m <<= 1)
                for (int j = 0; j < 8; ++j)
                    rs[j] += __shfl_xor(rs[j], m, 32);
            for (int j = 0; j < 8; ++j) l_i[j] += rs[j];

            u16* P = &Plds[w * 16 * 40];
            for (int j = 0; j < 8; ++j) {
                int prow = j + hbA;
                P[prow * 40 + r15]      = f2bf(s0.f[j]);
                P[prow * 40 + 16 + r15] = f2bf(s1.f[j]);
            }
            asm volatile("s_wait_dscnt 0" ::: "memory");
            FragA pa;
            pa.q[0] = *(const uint4*)&P[r15 * 40 + hbA];
            pa.q[1] = *(const uint4*)&P[r15 * 40 + 16 + hbA];

            for (int nt = 0; nt < 8; ++nt) {
                FragA fv;
                int col = nt * 16 + r15;
                for (int rr = 0; rr < 8; ++rr) {
                    int kk = hbB + rr * 2 + (rr >= 4 ? 8 : 0);
                    fv.w[rr] = *(const unsigned*)&Vt[col * 40 + kk];
                }
                acc[nt].v = __builtin_amdgcn_wmma_f32_16x16x32_bf16(false, pa.v, false, fv.v,
                                                                    (short)0, acc[nt].v, false, false);
            }
        }
        __syncthreads();
    }

    for (int nt = 0; nt < 8; ++nt)
        for (int j = 0; j < 8; ++j) {
            int q = q0 + j + hbA;
            int col = nt * 16 + r15;
            float v = acc[nt].f[j] / l_i[j];
            Y[((size_t)b * T_ + q) * HD_ + h * DH_ + col] = f2bf(v);
        }
}

// =====================================================================
__global__ __launch_bounds__(256) void rmsnorm_k(
    const float* __restrict__ X, const float* __restrict__ Wt,
    u16* __restrict__ OutH, float* __restrict__ OutF)
{
    __shared__ float red[8];
    const int row = blockIdx.x, tid = threadIdx.x;
    const float* xr = X + (size_t)row * C_;
    float vals[8], ss = 0.f;
    for (int i = 0; i < 8; ++i) { vals[i] = xr[tid + i * 256]; ss += vals[i] * vals[i]; }
    for (int m = 1; m < 32; m <<= 1) ss += __shfl_xor(ss, m, 32);
    if ((tid & 31) == 0) red[tid >> 5] = ss;
    __syncthreads();
    float tot = 0.f;
    for (int i = 0; i < 8; ++i) tot += red[i];
    float inv = rsqrtf(tot / (float)C_ + 1e-6f);
    for (int i = 0; i < 8; ++i) {
        int c = tid + i * 256;
        float o = Wt[c] * vals[i] * inv;
        if (OutH) OutH[(size_t)row * C_ + c] = f2bf(o);
        else      OutF[(size_t)row * C_ + c] = o;
    }
}

__global__ void rope_k(const int* __restrict__ pos,
                       float* __restrict__ Cos, float* __restrict__ Sin)
{
    int bt = blockIdx.x, j = threadIdx.x;
    int sel = 0;
    if (j < 60) { int m3 = j % 3; if (m3 == 1) sel = 1; else if (m3 == 2) sel = 2; }
    float p = (float)pos[(size_t)sel * BT_ + bt];
    float inv = expf(-((float)(2 * j) / (float)DH_) * logf(1000000.0f));
    float f = p * inv;
    Cos[(size_t)bt * 64 + j] = cosf(f);
    Sin[(size_t)bt * 64 + j] = sinf(f);
}

__global__ __launch_bounds__(128) void qkrope_k(
    const float* __restrict__ Qf, const float* __restrict__ Kf,
    const float* __restrict__ qnw, const float* __restrict__ knw,
    const float* __restrict__ Cos, const float* __restrict__ Sin,
    u16* __restrict__ Qb, u16* __restrict__ Kb)
{
    __shared__ float nv[128];
    __shared__ float red[4];
    const int rr = blockIdx.x, d = threadIdx.x;
    const int NQ = BT_ * H_;
    float v, wgt; size_t dst; int bt;
    if (rr < NQ) {
        bt = rr / H_; int hh = rr % H_;
        v = Qf[(size_t)bt * HD_ + hh * DH_ + d]; wgt = qnw[d];
        int b = bt / T_, t = bt % T_;
        dst = (((size_t)b * H_ + hh) * T_ + t) * DH_ + d;
    } else {
        int r2 = rr - NQ; bt = r2 / KV_; int kv = r2 % KV_;
        v = Kf[(size_t)bt * KVD_ + kv * DH_ + d]; wgt = knw[d];
        int b = bt / T_, t = bt % T_;
        dst = (((size_t)b * KV_ + kv) * T_ + t) * DH_ + d;
    }
    float ss = v * v;
    for (int m = 1; m < 32; m <<= 1) ss += __shfl_xor(ss, m, 32);
    if ((d & 31) == 0) red[d >> 5] = ss;
    __syncthreads();
    float tot = red[0] + red[1] + red[2] + red[3];
    float n = v * rsqrtf(tot / (float)DH_ + 1e-6f) * wgt;
    nv[d] = n;
    __syncthreads();
    int j = d & 63;
    float c = Cos[(size_t)bt * 64 + j], s = Sin[(size_t)bt * 64 + j];
    float rot = (d < 64) ? -nv[d + 64] : nv[d - 64];
    float o = n * c + rot * s;
    if (rr < NQ) Qb[dst] = f2bf(o); else Kb[dst] = f2bf(o);
}

__global__ void vconv_k(const float* __restrict__ Vf, u16* __restrict__ Vb)
{
    int i = blockIdx.x * 256 + threadIdx.x;
    int bt = i >> 9, c = i & 511;
    int kv = c >> 7, d = c & 127;
    int b = bt / T_, t = bt % T_;
    Vb[(((size_t)b * KV_ + kv) * T_ + t) * DH_ + d] = f2bf(Vf[i]);
}

__global__ void embed_k(const int* __restrict__ ids,
                        const float* __restrict__ Et, float* __restrict__ X)
{
    int bt = blockIdx.x;
    int id = ids[bt];
    for (int i = threadIdx.x; i < C_; i += 256)
        X[(size_t)bt * C_ + i] = Et[(size_t)id * C_ + i];
}

__global__ __launch_bounds__(256) void gate_k(
    const u16* __restrict__ Xn, const float* __restrict__ GW, float* __restrict__ Gate)
{
    __shared__ float sc[8];
    int bt = blockIdx.x, lane = threadIdx.x & 31, w = threadIdx.x >> 5;
    const u16* xr = Xn + (size_t)bt * C_;
    const float* gw = GW + (size_t)w * C_;
    float acc = 0.f;
    for (int d = lane; d < C_; d += 32) acc += bf2f(xr[d]) * gw[d];
    for (int m = 1; m < 32; m <<= 1) acc += __shfl_xor(acc, m, 32);
    if (lane == 0) sc[w] = acc;
    __syncthreads();
    if (threadIdx.x == 0) {
        int i1 = 0;
        for (int e = 1; e < E_; ++e) if (sc[e] > sc[i1]) i1 = e;
        int i2 = -1;
        for (int e = 0; e < E_; ++e) if (e != i1 && (i2 < 0 || sc[e] > sc[i2])) i2 = e;
        float e2 = expf(sc[i2] - sc[i1]);
        float p1 = 1.f / (1.f + e2);
        for (int e = 0; e < E_; ++e) Gate[(size_t)bt * E_ + e] = 0.f;
        Gate[(size_t)bt * E_ + i1] = p1;
        Gate[(size_t)bt * E_ + i2] = e2 * p1;
    }
}

// Deterministic per-expert token compaction (stable ascending order).
__global__ __launch_bounds__(256) void route_k(
    const float* __restrict__ Gate,
    int* __restrict__ cnt, int* __restrict__ idx, float* __restrict__ wsc)
{
    __shared__ int wtot[8];
    __shared__ int sbase;
    const int e = blockIdx.x;
    const int tid = threadIdx.x, lane = tid & 31, w = tid >> 5;
    if (tid == 0) sbase = 0;
    __syncthreads();
    for (int t0 = 0; t0 < BT_; t0 += 256) {
        int t = t0 + tid;
        float g = Gate[(size_t)t * E_ + e];
        bool sel = g > 0.f;
        unsigned mask = (unsigned)__ballot(sel);
        int pos = __popc(mask & ((1u << lane) - 1u));
        if (lane == 0) wtot[w] = __popc(mask);
        __syncthreads();
        int woff = 0, tot = 0;
        for (int i = 0; i < 8; ++i) { if (i < w) woff += wtot[i]; tot += wtot[i]; }
        int base = sbase;
        if (sel) {
            int slot = base + woff + pos;
            idx[(size_t)e * BT_ + slot] = t;
            wsc[(size_t)e * BT_ + slot] = g;
        }
        __syncthreads();
        if (tid == 0) sbase = base + tot;
        __syncthreads();
    }
    if (tid == 0) cnt[e] = sbase;
}

__global__ void silu_k(const float* __restrict__ G, const float* __restrict__ U,
                       u16* __restrict__ Hh, const int* __restrict__ cntPtr)
{
    int n = (*cntPtr) * M_;
    int i = blockIdx.x * 256 + threadIdx.x;
    if (i >= n) return;
    float g = G[i], u = U[i];
    Hh[i] = f2bf((g / (1.f + expf(-g))) * u);
}

// =====================================================================
extern "C" void kernel_launch(void* const* d_in, const int* in_sizes, int n_in,
                              void* d_out, int out_size, void* d_ws, size_t ws_size,
                              hipStream_t stream)
{
    const int*   ids   = (const int*)  d_in[0];
    const int*   pos   = (const int*)  d_in[1];
    const float* emb   = (const float*)d_in[2];
    const float* ln1w  = (const float*)d_in[3];
    const float* qwF   = (const float*)d_in[4];
    const float* kwF   = (const float*)d_in[5];
    const float* vwF   = (const float*)d_in[6];
    const float* owF   = (const float*)d_in[7];
    const float* qnw   = (const float*)d_in[8];
    const float* knw   = (const float*)d_in[9];
    const float* ln2w  = (const float*)d_in[10];
    const float* gw    = (const float*)d_in[11];
    const float* wgF   = (const float*)d_in[12];
    const float* wuF   = (const float*)d_in[13];
    const float* wdF   = (const float*)d_in[14];
    const float* normw = (const float*)d_in[15];

    char* p = (char*)d_ws;
    auto alloc = [&](size_t bytes) -> void* {
        void* r = (void*)p;
        p += (bytes + 255) & ~(size_t)255;
        return r;
    };
    float* x    = (float*)alloc((size_t)BT_ * C_ * 4);
    float* qf   = (float*)alloc((size_t)BT_ * HD_ * 4);
    float* kf   = (float*)alloc((size_t)BT_ * KVD_ * 4);
    float* vf   = (float*)alloc((size_t)BT_ * KVD_ * 4);
    float* gbuf = (float*)alloc((size_t)BT_ * M_ * 4);
    float* ubuf = (float*)alloc((size_t)BT_ * M_ * 4);
    float* gate = (float*)alloc((size_t)BT_ * E_ * 4);
    float* cosb = (float*)alloc((size_t)BT_ * 64 * 4);
    float* sinb = (float*)alloc((size_t)BT_ * 64 * 4);
    int*   cnt  = (int*)  alloc((size_t)E_ * 4);
    int*   idxb = (int*)  alloc((size_t)E_ * BT_ * 4);
    float* wscb = (float*)alloc((size_t)E_ * BT_ * 4);
    u16* xn  = (u16*)alloc((size_t)BT_ * C_ * 2);
    u16* Qb  = (u16*)alloc((size_t)BT_ * HD_ * 2);
    u16* Kb  = (u16*)alloc((size_t)BT_ * KVD_ * 2);
    u16* Vb  = (u16*)alloc((size_t)BT_ * KVD_ * 2);
    u16* ybf = (u16*)alloc((size_t)BT_ * HD_ * 2);
    u16* hbf = (u16*)alloc((size_t)BT_ * M_ * 2);
    const size_t nQW = (size_t)L_ * HD_ * C_;
    const size_t nKW = (size_t)L_ * KVD_ * C_;
    const size_t nOW = (size_t)L_ * C_ * HD_;
    const size_t nGU = (size_t)L_ * E_ * M_ * C_;
    const size_t nDW = (size_t)L_ * E_ * C_ * M_;
    u16* qw = (u16*)alloc(nQW * 2);
    u16* kw = (u16*)alloc(nKW * 2);
    u16* vw = (u16*)alloc(nKW * 2);
    u16* ow = (u16*)alloc(nOW * 2);
    u16* wg = (u16*)alloc(nGU * 2);
    u16* wu = (u16*)alloc(nGU * 2);
    u16* wd = (u16*)alloc(nDW * 2);

    auto cvt = [&](const float* s, u16* d, size_t n) {
        cvt_k<<<(unsigned)((n + 255) / 256), 256, 0, stream>>>(s, d, n);
    };
    cvt(qwF, qw, nQW); cvt(kwF, kw, nKW); cvt(vwF, vw, nKW); cvt(owF, ow, nOW);
    cvt(wgF, wg, nGU); cvt(wuF, wu, nGU); cvt(wdF, wd, nDW);

    auto gemm = [&](const u16* A, const u16* W, float* O,
                    int M, int N, int K, int beta,
                    const int* gIdx, const int* sIdx, const float* ssc, const int* cp) {
        gemm_bf16_wmma<<<dim3(N / 128, M / 64), 256, 0, stream>>>(
            A, W, O, M, N, K, beta, gIdx, sIdx, ssc, cp);
    };

    embed_k<<<BT_, 256, 0, stream>>>(ids, emb, x);
    rope_k<<<BT_, 64, 0, stream>>>(pos, cosb, sinb);

    for (int l = 0; l < L_; ++l) {
        // ---- attention block ----
        rmsnorm_k<<<BT_, 256, 0, stream>>>(x, ln1w + (size_t)l * C_, xn, nullptr);
        gemm(xn, qw + (size_t)l * HD_ * C_,  qf, BT_, HD_,  C_, 0, nullptr, nullptr, nullptr, nullptr);
        gemm(xn, kw + (size_t)l * KVD_ * C_, kf, BT_, KVD_, C_, 0, nullptr, nullptr, nullptr, nullptr);
        gemm(xn, vw + (size_t)l * KVD_ * C_, vf, BT_, KVD_, C_, 0, nullptr, nullptr, nullptr, nullptr);
        qkrope_k<<<BT_ * (H_ + KV_), 128, 0, stream>>>(
            qf, kf, qnw + (size_t)l * DH_, knw + (size_t)l * DH_, cosb, sinb, Qb, Kb);
        vconv_k<<<(BT_ * KVD_) / 256, 256, 0, stream>>>(vf, Vb);
        attn_kernel<<<dim3(T_ / 64, H_, B_), 128, 0, stream>>>(Qb, Kb, Vb, ybf);
        gemm(ybf, ow + (size_t)l * C_ * HD_, x, BT_, C_, HD_, 1, nullptr, nullptr, nullptr, nullptr);

        // ---- MoE block (top-2 routed, compacted per expert) ----
        rmsnorm_k<<<BT_, 256, 0, stream>>>(x, ln2w + (size_t)l * C_, xn, nullptr);
        gate_k<<<BT_, 256, 0, stream>>>(xn, gw + (size_t)l * E_ * C_, gate);
        route_k<<<E_, 256, 0, stream>>>(gate, cnt, idxb, wscb);
        for (int e = 0; e < E_; ++e) {
            const u16*   wge  = wg + ((size_t)l * E_ + e) * M_ * C_;
            const u16*   wue  = wu + ((size_t)l * E_ + e) * M_ * C_;
            const u16*   wde  = wd + ((size_t)l * E_ + e) * C_ * M_;
            const int*   cntE = cnt + e;
            const int*   idxE = idxb + (size_t)e * BT_;
            const float* wscE = wscb + (size_t)e * BT_;
            gemm(xn,  wge, gbuf, BT_, M_, C_, 0, idxE, nullptr, nullptr, cntE);
            gemm(xn,  wue, ubuf, BT_, M_, C_, 0, idxE, nullptr, nullptr, cntE);
            silu_k<<<(BT_ * M_) / 256, 256, 0, stream>>>(gbuf, ubuf, hbf, cntE);
            gemm(hbf, wde, x, BT_, C_, M_, 1, nullptr, idxE, wscE, cntE);
        }
    }
    rmsnorm_k<<<BT_, 256, 0, stream>>>(x, normw, nullptr, (float*)d_out);
}
